// GraphConv_53687091200297
// MI455X (gfx1250) — matrix operand order repaired and checked by reading, not compiled, not run on hardware
//
#include <hip/hip_runtime.h>

// GraphConv: out[b,v,:] = relu(concat_r(nodes[b, map[b,v,r], :]) @ kernel + bias)
// B=16, V=10000, R=8, C=64, UNITS=64  ->  GEMM M=160000, K=512, N=64
//
// bf16-split WMMA: x = hi + lo (both bf16); A*B ~= Ahi*Bhi + Alo*Bhi + Ahi*Blo
// using v_wmma_f32_16x16x32_bf16 (wave32, 16x16x32).
//
// Fast path pre-splits BOTH operands into bf16 hi/lo planes in d_ws so the
// inner loop is loads + WMMA only (no per-element convert VALU).

typedef __attribute__((ext_vector_type(16))) __bf16 v16bf;
typedef __attribute__((ext_vector_type(8)))  __bf16 v8bf;
typedef __attribute__((ext_vector_type(8)))  float  v8f;

#define B_  16
#define V_  10000
#define R_  8
#define C_  64
#define U_  64
#define K_  (R_ * C_)            // 512
#define CH_ ((V_ + 63) / 64)     // 157 chunks of 64 rows per batch
#define NODES_ELEMS (B_ * V_ * C_)          // 10,240,000

// workspace byte layout
#define KHI_OFF 0u                           // 64 KB  kernel hi, transposed [64][512]
#define KLO_OFF 65536u                       // 64 KB  kernel lo, transposed [64][512]
#define Z_OFF   131072u                      // 256 B  zeros (read as f32 or bf16)
#define NHI_OFF 131584u                      // 20.48 MB nodes hi plane [B*V][64]
#define NLO_OFF (NHI_OFF + 2u * NODES_ELEMS) // 20.48 MB nodes lo plane
#define WS_NEED ((size_t)NLO_OFF + 2u * NODES_ELEMS)

// ---------------------------------------------------------------------------
// Prep 1: split kernel[512][64] f32 into transposed bf16 planes [64][512],
// and zero the null row used for mapping == -1.
// ---------------------------------------------------------------------------
__global__ __launch_bounds__(256) void prep_kernel(
    const float* __restrict__ kern, char* __restrict__ wsb)
{
    __bf16* khiT = reinterpret_cast<__bf16*>(wsb + KHI_OFF);
    __bf16* kloT = reinterpret_cast<__bf16*>(wsb + KLO_OFF);
    float*  zrow = reinterpret_cast<float*>(wsb + Z_OFF);

    int t = blockIdx.x * blockDim.x + threadIdx.x;   // 0 .. 32767
    if (t < C_) zrow[t] = 0.0f;
    if (t < K_ * U_) {
        int f = t / U_;          // K index 0..511
        int u = t % U_;          // unit index 0..63
        float x = kern[t];       // kern[f][u]
        __bf16 h = (__bf16)x;
        khiT[(size_t)u * K_ + f] = h;
        kloT[(size_t)u * K_ + f] = (__bf16)(x - (float)h);
    }
}

// ---------------------------------------------------------------------------
// Prep 2 (fast path): split nodes into bf16 hi/lo planes, row layout kept.
// ---------------------------------------------------------------------------
__global__ __launch_bounds__(256) void prep_nodes(
    const float* __restrict__ nodes, char* __restrict__ wsb)
{
    __bf16* nhi = reinterpret_cast<__bf16*>(wsb + NHI_OFF);
    __bf16* nlo = reinterpret_cast<__bf16*>(wsb + NLO_OFF);
    int t = blockIdx.x * blockDim.x + threadIdx.x;   // grid sized exactly
    float x = nodes[t];
    __bf16 h = (__bf16)x;
    nhi[t] = h;
    nlo[t] = (__bf16)(x - (float)h);
}

// ---------------------------------------------------------------------------
// Shared store: C/D layout has VGPR j holding M = j + 8*g, N = m.
// ---------------------------------------------------------------------------
__device__ __forceinline__ void store_tile(
    v8f acc[4][4], const float* __restrict__ bias, float* __restrict__ outB,
    int v0, int m, int g)
{
    float brow[4];
    #pragma unroll
    for (int nt = 0; nt < 4; ++nt) brow[nt] = bias[nt * 16 + m];

    #pragma unroll
    for (int mt = 0; mt < 4; ++mt) {
        #pragma unroll
        for (int j = 0; j < 8; ++j) {
            const int v = v0 + mt * 16 + g * 8 + j;
            if (v < V_) {
                float* orow = outB + (size_t)v * U_;
                #pragma unroll
                for (int nt = 0; nt < 4; ++nt) {
                    float val = acc[mt][nt][j] + brow[nt];
                    __builtin_nontemporal_store(val > 0.f ? val : 0.f,
                                                &orow[nt * 16 + m]);
                }
            }
        }
    }
}

// ---------------------------------------------------------------------------
// Fast path: everything pre-split to bf16; inner loop = loads + WMMA.
// One wave32 computes a 64(rows) x 64(units) output tile.
// ---------------------------------------------------------------------------
__global__ __launch_bounds__(256) void graphconv_wmma_pre(
    const char* __restrict__ wsb,
    const int*  __restrict__ mapping,
    const float* __restrict__ bias,
    float* __restrict__ out)
{
    const int lane = threadIdx.x & 31;
    const int m    = lane & 15;
    const int g    = lane >> 4;
    const int wid  = blockIdx.x * (blockDim.x >> 5) + (threadIdx.x >> 5);
    const int b    = wid / CH_;
    const int v0   = (wid % CH_) * 64;
    if (b >= B_) return;                      // wave-uniform

    float* outB = out + (size_t)b * V_ * U_;

    const unsigned baseH = NHI_OFF + (unsigned)(b * V_) * (C_ * 2u);
    const unsigned baseL = NLO_OFF + (unsigned)(b * V_) * (C_ * 2u);
    const unsigned g16   = (unsigned)g * 32u; // lane-half byte offset (K grouping)

    const int* mapRow[4];
    #pragma unroll
    for (int mt = 0; mt < 4; ++mt) {
        int v  = v0 + mt * 16 + m;
        int vc = v < V_ ? v : V_ - 1;
        mapRow[mt] = mapping + ((size_t)b * V_ + vc) * R_;
    }

    const v8f zacc = {0.f, 0.f, 0.f, 0.f, 0.f, 0.f, 0.f, 0.f};
    v8f acc[4][4];
    #pragma unroll
    for (int mt = 0; mt < 4; ++mt)
        #pragma unroll
        for (int nt = 0; nt < 4; ++nt)
            acc[mt][nt] = zacc;

    #pragma unroll 1
    for (int r = 0; r < R_; ++r) {
        // gathered row byte-offsets into the hi/lo planes (-1 -> zero row)
        unsigned hoff[4], loff[4];
        #pragma unroll
        for (int mt = 0; mt < 4; ++mt) {
            int i = mapRow[mt][r];
            unsigned o = (unsigned)i * (C_ * 2u);
            hoff[mt] = (i >= 0) ? (baseH + o + g16) : (Z_OFF + g16);
            loff[mt] = (i >= 0) ? (baseL + o + g16) : (Z_OFF + g16);
        }

        #pragma unroll
        for (int half = 0; half < 2; ++half) {
            const int s   = r * 2 + half;
            const int cb2 = half * 64;            // byte offset of K-half in row
            const int kb2 = (s * 32 + g * 16) * 2; // B-operand K byte base

            v16bf bh[4], bl[4];
            #pragma unroll
            for (int nt = 0; nt < 4; ++nt) {
                const size_t col = (size_t)(nt * 16 + m) * (K_ * 2);
                bh[nt] = *reinterpret_cast<const v16bf*>(wsb + KHI_OFF + col + kb2);
                bl[nt] = *reinterpret_cast<const v16bf*>(wsb + KLO_OFF + col + kb2);
            }

            #pragma unroll
            for (int mt = 0; mt < 4; ++mt) {
                // A in this code's contiguous k order (B matches): lane half g holds
                //   K = cb + g*16 + [0..15]
                const char* hp = wsb + hoff[mt] + cb2;
                const char* lp = wsb + loff[mt] + cb2;
                v8bf h1 = *reinterpret_cast<const v8bf*>(hp);
                v8bf h2 = *reinterpret_cast<const v8bf*>(hp + 16);
                v8bf l1 = *reinterpret_cast<const v8bf*>(lp);
                v8bf l2 = *reinterpret_cast<const v8bf*>(lp + 16);
                v16bf ah = __builtin_shufflevector(h1, h2, 0, 1, 2, 3, 4, 5, 6, 7,
                                                   8, 9, 10, 11, 12, 13, 14, 15);
                v16bf al = __builtin_shufflevector(l1, l2, 0, 1, 2, 3, 4, 5, 6, 7,
                                                   8, 9, 10, 11, 12, 13, 14, 15);

                #pragma unroll
                for (int nt = 0; nt < 4; ++nt) {
                    acc[mt][nt] = __builtin_amdgcn_wmma_f32_16x16x32_bf16(
                        false, ah, false, bh[nt], (short)0, acc[mt][nt], false, false);
                    acc[mt][nt] = __builtin_amdgcn_wmma_f32_16x16x32_bf16(
                        false, al, false, bh[nt], (short)0, acc[mt][nt], false, false);
                    acc[mt][nt] = __builtin_amdgcn_wmma_f32_16x16x32_bf16(
                        false, ah, false, bl[nt], (short)0, acc[mt][nt], false, false);
                }
            }
        }
    }

    store_tile(acc, bias, outB, v0, m, g);
}

// ---------------------------------------------------------------------------
// Fallback path (small ws): split A in-loop from f32 nodes.
// ---------------------------------------------------------------------------
__global__ __launch_bounds__(256) void graphconv_wmma(
    const float* __restrict__ nodes,
    const int*   __restrict__ mapping,
    const float* __restrict__ bias,
    const char*  __restrict__ wsb,
    float* __restrict__ out)
{
    const __bf16* khiT = reinterpret_cast<const __bf16*>(wsb + KHI_OFF);
    const __bf16* kloT = reinterpret_cast<const __bf16*>(wsb + KLO_OFF);
    const float*  zrow = reinterpret_cast<const float*>(wsb + Z_OFF);

    const int lane = threadIdx.x & 31;
    const int m    = lane & 15;
    const int g    = lane >> 4;
    const int wid  = blockIdx.x * (blockDim.x >> 5) + (threadIdx.x >> 5);
    const int b    = wid / CH_;
    const int v0   = (wid % CH_) * 64;
    if (b >= B_) return;

    const float* nodesB = nodes + (size_t)b * V_ * C_;
    float*       outB   = out   + (size_t)b * V_ * U_;

    const int* mapRow[4];
    #pragma unroll
    for (int mt = 0; mt < 4; ++mt) {
        int v  = v0 + mt * 16 + m;
        int vc = v < V_ ? v : V_ - 1;
        mapRow[mt] = mapping + ((size_t)b * V_ + vc) * R_;
    }

    const v8f zacc = {0.f, 0.f, 0.f, 0.f, 0.f, 0.f, 0.f, 0.f};
    v8f acc[4][4];
    #pragma unroll
    for (int mt = 0; mt < 4; ++mt)
        #pragma unroll
        for (int nt = 0; nt < 4; ++nt)
            acc[mt][nt] = zacc;

    #pragma unroll 1
    for (int r = 0; r < R_; ++r) {
        int idx[4];
        #pragma unroll
        for (int mt = 0; mt < 4; ++mt) idx[mt] = mapRow[mt][r];

        #pragma unroll
        for (int half = 0; half < 2; ++half) {
            const int s  = r * 2 + half;
            const int cb = half * 32;
            const int kb = s * 32 + g * 16;

            v16bf bh[4], bl[4];
            #pragma unroll
            for (int nt = 0; nt < 4; ++nt) {
                const int col = nt * 16 + m;
                bh[nt] = *reinterpret_cast<const v16bf*>(khiT + (size_t)col * K_ + kb);
                bl[nt] = *reinterpret_cast<const v16bf*>(kloT + (size_t)col * K_ + kb);
            }

            #pragma unroll
            for (int mt = 0; mt < 4; ++mt) {
                const float* src = (idx[mt] >= 0)
                                       ? (nodesB + (size_t)idx[mt] * C_)
                                       : zrow;
                const float4 c10 = *reinterpret_cast<const float4*>(src + cb + g * 16);
                const float4 c11 = *reinterpret_cast<const float4*>(src + cb + g * 16 + 4);
                const float4 c20 = *reinterpret_cast<const float4*>(src + cb + g * 16 + 8);
                const float4 c21 = *reinterpret_cast<const float4*>(src + cb + g * 16 + 12);

                const float xs[16] = {c10.x, c10.y, c10.z, c10.w,
                                      c11.x, c11.y, c11.z, c11.w,
                                      c20.x, c20.y, c20.z, c20.w,
                                      c21.x, c21.y, c21.z, c21.w};
                v16bf ah, al;
                #pragma unroll
                for (int e = 0; e < 16; ++e) {
                    __bf16 h = (__bf16)xs[e];
                    ah[e] = h;
                    al[e] = (__bf16)(xs[e] - (float)h);
                }

                #pragma unroll
                for (int nt = 0; nt < 4; ++nt) {
                    acc[mt][nt] = __builtin_amdgcn_wmma_f32_16x16x32_bf16(
                        false, ah, false, bh[nt], (short)0, acc[mt][nt], false, false);
                    acc[mt][nt] = __builtin_amdgcn_wmma_f32_16x16x32_bf16(
                        false, al, false, bh[nt], (short)0, acc[mt][nt], false, false);
                    acc[mt][nt] = __builtin_amdgcn_wmma_f32_16x16x32_bf16(
                        false, ah, false, bl[nt], (short)0, acc[mt][nt], false, false);
                }
            }
        }
    }

    store_tile(acc, bias, outB, v0, m, g);
}

// ---------------------------------------------------------------------------
extern "C" void kernel_launch(void* const* d_in, const int* in_sizes, int n_in,
                              void* d_out, int out_size, void* d_ws, size_t ws_size,
                              hipStream_t stream) {
    const float* nodes   = (const float*)d_in[0];   // [B, V, C] f32
    const int*   mapping = (const int*)d_in[1];     // [B, V, R] i32
    const float* kern    = (const float*)d_in[2];   // [R*C, U] f32
    const float* bias    = (const float*)d_in[3];   // [U] f32
    float* out = (float*)d_out;                     // [B, V, U] f32
    char*  wsb = (char*)d_ws;

    prep_kernel<<<(K_ * U_) / 256, 256, 0, stream>>>(kern, wsb);

    const int total_waves = B_ * CH_;               // 2512
    if (ws_size >= WS_NEED) {
        prep_nodes<<<NODES_ELEMS / 256, 256, 0, stream>>>(nodes, wsb);
        graphconv_wmma_pre<<<total_waves / 8, 256, 0, stream>>>(
            wsb, mapping, bias, out);
    } else {
        graphconv_wmma<<<total_waves / 8, 256, 0, stream>>>(
            nodes, mapping, bias, wsb, out);
    }
}
